// QBlock_79173427135012
// MI455X (gfx1250) — 8-bit matrix operand order repaired and checked by reading, compiled, not run on hardware
//
#include <hip/hip_runtime.h>
#include <hip/hip_bf16.h>

typedef int v8i __attribute__((ext_vector_type(8)));

constexpr int    CH  = 64;
constexpr int    HW  = 65536;
constexpr int    NB  = 16;
constexpr size_t TOT = (size_t)NB * CH * HW;

#if defined(__has_builtin)
#if __has_builtin(__builtin_amdgcn_global_load_async_to_lds_b32)
#define HAVE_ASYNC_LDS 1
#endif
#endif
#ifndef HAVE_ASYNC_LDS
#define HAVE_ASYNC_LDS 0
#endif

typedef __attribute__((address_space(1))) int as1_int;
typedef __attribute__((address_space(3))) int as3_int;


__global__ void init_slots(unsigned* slots) {
    if (threadIdx.x < 4) slots[threadIdx.x] = 0u;
}

__global__ void absmax_kernel(const float* __restrict__ x, size_t n4, unsigned* out) {
    __shared__ float red[256];
    float m = 0.f;
    const float4* p = (const float4*)x;
    for (size_t i = (size_t)blockIdx.x * blockDim.x + threadIdx.x; i < n4;
         i += (size_t)gridDim.x * blockDim.x) {
        float4 v = p[i];
        m = fmaxf(m, fmaxf(fmaxf(fabsf(v.x), fabsf(v.y)), fmaxf(fabsf(v.z), fabsf(v.w))));
    }
    red[threadIdx.x] = m;
    __syncthreads();
    for (int s = 128; s > 0; s >>= 1) {
        if (threadIdx.x < s) red[threadIdx.x] = fmaxf(red[threadIdx.x], red[threadIdx.x + s]);
        __syncthreads();
    }
    if (threadIdx.x == 0) atomicMax(out, __float_as_uint(red[0]));
}

__device__ float block_absmax256(const float* p, int n, float* red) {
    float m = 0.f;
    for (int i = threadIdx.x; i < n; i += 256) m = fmaxf(m, fabsf(p[i]));
    red[threadIdx.x] = m;
    __syncthreads();
    for (int s = 128; s > 0; s >>= 1) {
        if (threadIdx.x < s) red[threadIdx.x] = fmaxf(red[threadIdx.x], red[threadIdx.x + s]);
        __syncthreads();
    }
    float r = red[0];
    __syncthreads();
    return r;
}

__global__ void prep_weights(const float* __restrict__ wp1, const float* __restrict__ wf1,
                             const float* __restrict__ wp2, const float* __restrict__ wf2,
                             float* __restrict__ scales, int* __restrict__ ap1,
                             int* __restrict__ ap2, float* __restrict__ wf1dq,
                             float* __restrict__ wf2dq) {
    __shared__ float red[256];
    const float s1 = block_absmax256(wp1, CH * CH, red) * (1.f / 7.f) + 1e-12f;
    const float s2 = block_absmax256(wf1, CH * 9, red) * (1.f / 7.f) + 1e-12f;
    const float s3 = block_absmax256(wp2, CH * CH, red) * (1.f / 7.f) + 1e-12f;
    const float s4 = block_absmax256(wf2, CH * 9, red) * (1.f / 7.f) + 1e-12f;
    if (threadIdx.x == 0) { scales[0] = s1; scales[1] = s2; scales[2] = s3; scales[3] = s4; }

    for (int i = threadIdx.x; i < 1024; i += 256) {
        const int tile = i >> 8, lane = (i >> 3) & 31, v = i & 7;
        const int M   = (tile << 4) + (lane & 15);
        const int hiK = (lane >> 4) << 3;
        int pk1 = 0, pk2 = 0;
#pragma unroll
        for (int b = 0; b < 4; ++b) {
            const int K  = ((v >> 1) << 4) + ((v & 1) << 2) + hiK + b;
            float q1 = fminf(fmaxf(rintf(wp1[M * CH + K] / s1), -7.f), 7.f);
            float q2 = fminf(fmaxf(rintf(wp2[M * CH + K] / s3), -7.f), 7.f);
            pk1 |= ((int)q1 & 0xff) << (8 * b);
            pk2 |= ((int)q2 & 0xff) << (8 * b);
        }
        ap1[i] = pk1;
        ap2[i] = pk2;
    }
    for (int i = threadIdx.x; i < CH * 9; i += 256) {
        wf1dq[i] = fminf(fmaxf(rintf(wf1[i] / s2), -7.f), 7.f) * s2;
        wf2dq[i] = fminf(fmaxf(rintf(wf2[i] / s4), -7.f), 7.f) * s4;
    }
}

__global__ void pw_conv_wmma(const float* __restrict__ x, const int* __restrict__ apack,
                             const unsigned* __restrict__ amax_in,
                             const float* __restrict__ wscale, float* __restrict__ y,
                             unsigned* __restrict__ amax_out) {
    const int lane = threadIdx.x & 31;
    const int wv   = threadIdx.x >> 5;
    const int pxg  = blockIdx.x * 128 + wv * 16;
    const int n    = pxg >> 16;
    const int hw   = pxg & (HW - 1);
    const int px   = hw + (lane & 15);
    const int hi   = lane >> 4;

    const float s_a    = __uint_as_float(*amax_in) * (1.f / 127.f) + 1e-12f;
    const float inv_sa = 1.f / s_a;
    const float dq     = s_a * (*wscale);

    const float* xb = x + (size_t)n * CH * HW + px;
    __builtin_prefetch(xb + 128, 0, 1);

    v8i bf;
#pragma unroll
    for (int v = 0; v < 8; ++v) {
        int pk = 0;
#pragma unroll
        for (int b = 0; b < 4; ++b) {
            const int K = ((v >> 1) << 4) + ((v & 1) << 2) + (hi << 3) + b;
            float f  = xb[(size_t)K * HW];
            float qf = fminf(fmaxf(rintf(f * inv_sa), -127.f), 127.f);
            pk |= ((int)qf & 0xff) << (8 * b);
        }
        bf[v] = pk;
    }

    float lmax = 0.f;
#pragma unroll
    for (int t = 0; t < 4; ++t) {
        v8i af  = *(const v8i*)(apack + ((t << 5) + lane) * 8);
        v8i acc = {};
        acc = __builtin_amdgcn_wmma_i32_16x16x64_iu8(true, af, true, bf, acc, false, false);
#pragma unroll
        for (int r = 0; r < 8; ++r) {
            const int co = (t << 4) + r + (hi << 3);
            float val = (float)acc[r] * dq;
            y[(size_t)(n * CH + co) * HW + px] = val;
            lmax = fmaxf(lmax, fabsf(val));
        }
    }
#pragma unroll
    for (int off = 16; off; off >>= 1) lmax = fmaxf(lmax, __shfl_xor(lmax, off, 32));
    if (lane == 0) atomicMax(amax_out, __float_as_uint(lmax));
}

__global__ void dw_prelu(const float* __restrict__ in, const unsigned* __restrict__ amax_in,
                         const float* __restrict__ wdq, const float* __restrict__ alpha,
                         const float* __restrict__ residual, float* __restrict__ out,
                         unsigned* amax_out) {
    __shared__ float sm[10 * 256];

    const int strip = blockIdx.x & 31;
    const int plane = blockIdx.x >> 5;
    const int c     = plane & (CH - 1);
    const int h0    = strip * 8;
    const int w     = threadIdx.x;
    const size_t base = (size_t)plane * HW;

#pragma unroll
    for (int lr = 0; lr < 10; ++lr) {
        const int h = h0 - 1 + lr;
        if ((unsigned)h < 256u) {
#if HAVE_ASYNC_LDS
            const float* gp = in + base + h * 256 + w;
            as1_int* g = (as1_int*)(uintptr_t)gp;
            as3_int* l = (as3_int*)(unsigned)(uintptr_t)&sm[lr * 256 + w];
            __builtin_amdgcn_global_load_async_to_lds_b32(g, l, 0, 0);
#else
            sm[lr * 256 + w] = in[base + h * 256 + w];
#endif
        } else {
            sm[lr * 256 + w] = 0.f;
        }
    }
#if HAVE_ASYNC_LDS
#if defined(__has_builtin) && __has_builtin(__builtin_amdgcn_s_wait_asynccnt)
    __builtin_amdgcn_s_wait_asynccnt(0);
#else
    asm volatile("s_wait_asynccnt 0" ::: "memory");
#endif
#endif
    __syncthreads();

    const float s_a    = __uint_as_float(*amax_in) * (1.f / 127.f) + 1e-12f;
    const float inv_sa = 1.f / s_a;
#pragma unroll
    for (int lr = 0; lr < 10; ++lr) {
        float v = sm[lr * 256 + w];
        sm[lr * 256 + w] = fminf(fmaxf(rintf(v * inv_sa), -127.f), 127.f) * s_a;
    }
    __syncthreads();

    const float w00 = wdq[c * 9 + 0], w01 = wdq[c * 9 + 1], w02 = wdq[c * 9 + 2];
    const float w10 = wdq[c * 9 + 3], w11 = wdq[c * 9 + 4], w12 = wdq[c * 9 + 5];
    const float w20 = wdq[c * 9 + 6], w21 = wdq[c * 9 + 7], w22 = wdq[c * 9 + 8];
    const float al  = alpha[c];
    const bool lok = (w > 0), rok = (w < 255);

    float lmax = 0.f;
#pragma unroll
    for (int r = 0; r < 8; ++r) {
        const float* r0 = &sm[r * 256 + w];
        const float* r1 = r0 + 256;
        const float* r2 = r1 + 256;
        float acc = (lok ? r0[-1] : 0.f) * w00 + r0[0] * w01 + (rok ? r0[1] : 0.f) * w02 +
                    (lok ? r1[-1] : 0.f) * w10 + r1[0] * w11 + (rok ? r1[1] : 0.f) * w12 +
                    (lok ? r2[-1] : 0.f) * w20 + r2[0] * w21 + (rok ? r2[1] : 0.f) * w22;
        float rv = acc > 0.f ? acc : al * acc;
        const size_t oidx = base + (size_t)(h0 + r) * 256 + w;
        if (residual) rv += residual[oidx];
        out[oidx] = rv;
        lmax = fmaxf(lmax, fabsf(rv));
    }

    if (amax_out) {
#pragma unroll
        for (int off = 16; off; off >>= 1) lmax = fmaxf(lmax, __shfl_xor(lmax, off, 32));
        if ((threadIdx.x & 31) == 0) atomicMax(amax_out, __float_as_uint(lmax));
    }
}

extern "C" void kernel_launch(void* const* d_in, const int* in_sizes, int n_in,
                              void* d_out, int out_size, void* d_ws, size_t ws_size,
                              hipStream_t stream) {
    const float* x   = (const float*)d_in[0];
    const float* wp1 = (const float*)d_in[1];
    const float* wf1 = (const float*)d_in[2];
    const float* wp2 = (const float*)d_in[3];
    const float* wf2 = (const float*)d_in[4];
    const float* a1  = (const float*)d_in[5];
    const float* a2  = (const float*)d_in[6];

    char*     ws     = (char*)d_ws;
    unsigned* slots  = (unsigned*)ws;
    float*    scales = (float*)(ws + 16);
    int*      ap1    = (int*)(ws + 1024);
    int*      ap2    = (int*)(ws + 5120);
    float*    wf1dq  = (float*)(ws + 9216);
    float*    wf2dq  = (float*)(ws + 11520);
    float*    big    = (float*)(ws + 16384);
    float*    out    = (float*)d_out;

    const int pwBlocks = (int)((TOT / CH) / 128);
    const int dwBlocks = NB * CH * 32;

    init_slots<<<1, 32, 0, stream>>>(slots);
    absmax_kernel<<<4096, 256, 0, stream>>>(x, TOT / 4, slots + 0);
    prep_weights<<<1, 256, 0, stream>>>(wp1, wf1, wp2, wf2, scales, ap1, ap2, wf1dq, wf2dq);

    pw_conv_wmma<<<pwBlocks, 256, 0, stream>>>(x, ap1, slots + 0, scales + 0, big, slots + 1);
    dw_prelu<<<dwBlocks, 256, 0, stream>>>(big, slots + 1, wf1dq, a1, nullptr, out, slots + 2);
    pw_conv_wmma<<<pwBlocks, 256, 0, stream>>>(out, ap2, slots + 2, scales + 2, big, slots + 3);
    dw_prelu<<<dwBlocks, 256, 0, stream>>>(big, slots + 3, wf2dq, a2, x, out, nullptr);
}
